// UniTRBlock_6425271075228
// MI455X (gfx1250) — matrix operand order repaired and checked by reading, not compiled, not run on hardware
//
#include <hip/hip_runtime.h>
#include <hip/hip_bf16.h>

// ---------------------------------------------------------------------------
// Problem constants (match reference)
// ---------------------------------------------------------------------------
#define NV 90000      // total voxels
#define VNUM 30000    // lidar voxel count (16-aligned: 1875*16)
#define DM 256        // d_model
#define NH 8          // heads
#define HDIM 32       // head dim
#define DFFN 2048     // ffn dim
#define NSETS 1024    // sets
#define KSET 90       // set size
#define KPAD 96       // padded set size (mult of 16)
#define MROWS (NSETS * KPAD)   // 98304 padded attention rows
#define LNEPS 1e-5f

typedef __attribute__((ext_vector_type(16))) __bf16 v16bf;
typedef __attribute__((ext_vector_type(8)))  __bf16 bf16x8;
typedef __attribute__((ext_vector_type(8)))  float  v8f;

// ---------------------------------------------------------------------------
// f32 -> bf16 conversion (weights + activations)
// ---------------------------------------------------------------------------
__global__ void cvt_bf16_kernel(const float* __restrict__ in, __bf16* __restrict__ out, int n) {
    int i = blockIdx.x * blockDim.x + threadIdx.x;
    if (i < n) out[i] = (__bf16)in[i];
}

// ---------------------------------------------------------------------------
// perm = segment_min(arange, flat_inds): first occurrence of each voxel
// ---------------------------------------------------------------------------
__global__ void perm_init_kernel(int* __restrict__ perm, int n) {
    int i = blockIdx.x * blockDim.x + threadIdx.x;
    if (i < n) perm[i] = 0x7FFFFFFF;
}
__global__ void perm_min_kernel(const int* __restrict__ flat, int* __restrict__ perm, int n) {
    int j = blockIdx.x * blockDim.x + threadIdx.x;
    if (j < n) atomicMin(&perm[flat[j]], j);
}

// ---------------------------------------------------------------------------
// Gather: qk = bf16(src[i]+pos[i]), feat = bf16(src[i]); padded rows = 0
// ---------------------------------------------------------------------------
__global__ __launch_bounds__(256) void gather_kernel(
    const float* __restrict__ src, const float* __restrict__ pos,
    const int* __restrict__ inds, __bf16* __restrict__ qk, __bf16* __restrict__ feat) {
    int r = blockIdx.x;
    int d = threadIdx.x;
    int s = r / KPAD, kk = r % KPAD;
    size_t o = (size_t)r * DM + d;
    if (kk < KSET) {
        int i = inds[s * KSET + kk];
        float sv = src[(size_t)i * DM + d];
        qk[o]   = (__bf16)(sv + pos[(size_t)i * DM + d]);
        feat[o] = (__bf16)sv;
    } else {
        qk[o]   = (__bf16)0.f;
        feat[o] = (__bf16)0.f;
    }
}

// ---------------------------------------------------------------------------
// Generic WMMA GEMM: out[M x Nout] = act(A[M x Kd] @ W[Nout x Kd]^T + bias (+addend))
// A, W bf16 row-major; bias f32; out f32 or bf16.
//
// Block = 128 threads (4 M-waves). Each K-step of 64 stages the shared B panel
// (64 cols x 64 K, 8KB) into LDS via global_load_async_to_lds_b128 (ASYNCcnt),
// then all 4 waves consume it with ds reads in WMMA B lane layout. A streams
// from global per wave. ~32 FLOP per global byte.
// ---------------------------------------------------------------------------
template<bool RELU, bool OUTBF, bool ADD>
__global__ __launch_bounds__(128) void gemm_wmma_kernel(
    const __bf16* __restrict__ A, const __bf16* __restrict__ W,
    const float* __restrict__ bias, const float* __restrict__ addend,
    void* __restrict__ outv, int Mtiles, int Kd, int Nout) {
    __shared__ __bf16 bpanel[64 * 64];   // [n_local(64)][k_local(64)] row-major, 8KB

    const int tid  = threadIdx.x;
    const int lane = tid & 31;
    const int wave = tid >> 5;
    const int mt   = blockIdx.x * 4 + wave;
    const bool active = (mt < Mtiles);
    const int n0   = blockIdx.y * 64;
    const int bl   = lane & 15;
    const int hi16 = lane >> 4;           // 0 or 1
    const int akoff = hi16 << 4;          // A lane K-base: 0 or 16
    const int bkoff = hi16 << 4;          // B lane K-base: 0 or 16

    const __bf16* arowp = A + (size_t)((active ? mt : 0) * 16 + bl) * Kd;
    const unsigned lds_base = (unsigned)(unsigned long long)(void*)&bpanel[0];

    v8f acc[4];
#pragma unroll
    for (int t = 0; t < 4; ++t)
#pragma unroll
        for (int e = 0; e < 8; ++e) acc[t][e] = 0.f;

    for (int kb = 0; kb < Kd; kb += 64) {
        if (kb) __syncthreads();   // previous panel fully consumed before overwrite

        // Stage B panel: 64 rows (W rows n0..n0+63) x 128 bytes = 512 x 16B chunks.
        // 128 threads -> 4 async 16B loads each, tracked by ASYNCcnt.
#pragma unroll
        for (int c = 0; c < 4; ++c) {
            const int chunk = tid + c * 128;          // 0..511
            const int r = chunk >> 3;                 // panel row (n-local)
            const int q = chunk & 7;                  // 16B chunk (of 8) within 128B row
            const unsigned voff = (unsigned)(((n0 + r) * Kd + kb) * 2 + q * 16);
            const unsigned dst  = lds_base + (unsigned)((r << 7) + (q << 4));
            asm volatile("global_load_async_to_lds_b128 %0, %1, %2"
                         :: "v"(dst), "v"(voff), "s"(W) : "memory");
        }
        asm volatile("s_wait_asynccnt 0x0" ::: "memory");
        __syncthreads();

        if (active) {
#pragma unroll
            for (int kc = 0; kc < 64; kc += 32) {
                // A tile 16x32: lane holds row (mt*16 + bl), K = {base..base+15}
                bf16x8 lo = *(const bf16x8*)(arowp + kb + kc + akoff);
                bf16x8 hi = *(const bf16x8*)(arowp + kb + kc + akoff + 8);
                v16bf a;
#pragma unroll
                for (int e = 0; e < 8; ++e) { a[e] = lo[e]; a[8 + e] = hi[e]; }
#pragma unroll
                for (int t = 0; t < 4; ++t) {
                    // B tile 32x16: lane holds col (t*16+bl), 16 contiguous K from LDS panel row
                    v16bf b = *(const v16bf*)&bpanel[(t * 16 + bl) * 64 + kc + bkoff];
                    acc[t] = __builtin_amdgcn_wmma_f32_16x16x32_bf16(
                        false, a, false, b, (short)0, acc[t], false, false);
                }
            }
        }
    }

    if (!active) return;

    // C/D layout: VGPR j -> row j (lanes 0-15) / row j+8 (lanes 16-31), col = lane&15
    const int rbase = mt * 16 + (hi16 ? 8 : 0);
#pragma unroll
    for (int t = 0; t < 4; ++t) {
        const int col = n0 + t * 16 + bl;
        const float bv = bias[col];
#pragma unroll
        for (int j = 0; j < 8; ++j) {
            const int row = rbase + j;
            float v = acc[t][j] + bv;
            if (ADD) v += addend[(size_t)row * Nout + col];
            if (RELU) v = fmaxf(v, 0.f);
            if (OUTBF) ((__bf16*)outv)[(size_t)row * Nout + col] = (__bf16)v;
            else       ((float*)outv)[(size_t)row * Nout + col] = v;
        }
    }
}

// ---------------------------------------------------------------------------
// Attention: one block per (set, head). K/V head slices staged in LDS.
// qkp: (MROWS, 512) bf16 [Q | K]; vp: (MROWS, 256) bf16; out: (MROWS, 256) bf16
// ---------------------------------------------------------------------------
__global__ __launch_bounds__(96) void attention_kernel(
    const __bf16* __restrict__ qkp, const __bf16* __restrict__ vp,
    const unsigned char* __restrict__ mask, __bf16* __restrict__ attn) {
    __shared__ __bf16 ks[KPAD * HDIM];
    __shared__ __bf16 vs[KPAD * HDIM];
    __shared__ unsigned char ms[KPAD];

    const int s = blockIdx.x, h = blockIdx.y, q = threadIdx.x;
    const size_t rb = (size_t)s * KPAD;

    if (q < KSET) {
        const __bf16* kr = qkp + (rb + q) * 512 + 256 + h * HDIM;
        const __bf16* vr = vp  + (rb + q) * DM  + h * HDIM;
#pragma unroll 4
        for (int d = 0; d < HDIM; ++d) { ks[q * HDIM + d] = kr[d]; vs[q * HDIM + d] = vr[d]; }
        ms[q] = mask[s * KSET + q];
    }
    __syncthreads();

    __bf16* orow = attn + (rb + q) * DM + h * HDIM;
    if (q >= KSET) {  // padded rows: zero so the out-proj GEMM reads clean data
#pragma unroll
        for (int d = 0; d < HDIM; ++d) orow[d] = (__bf16)0.f;
        return;
    }

    float qf[HDIM];
    const __bf16* qr = qkp + (rb + q) * 512 + h * HDIM;
#pragma unroll
    for (int d = 0; d < HDIM; ++d) qf[d] = (float)qr[d];

    const float scale = 0.17677669529663687f;  // 1/sqrt(32)
    float sc[KSET];
    float mx = -3.0e38f;
    for (int k = 0; k < KSET; ++k) {
        if (ms[k]) { sc[k] = -3.0e38f; continue; }
        float dot = 0.f;
#pragma unroll 8
        for (int d = 0; d < HDIM; ++d) dot += qf[d] * (float)ks[k * HDIM + d];
        sc[k] = dot * scale;
        mx = fmaxf(mx, sc[k]);
    }
    float denom = 0.f;
    float av[HDIM];
#pragma unroll
    for (int d = 0; d < HDIM; ++d) av[d] = 0.f;
    for (int k = 0; k < KSET; ++k) {
        float w = __expf(sc[k] - mx);
        denom += w;
#pragma unroll 8
        for (int d = 0; d < HDIM; ++d) av[d] += w * (float)vs[k * HDIM + d];
    }
    float inv = 1.f / denom;
#pragma unroll
    for (int d = 0; d < HDIM; ++d) orow[d] = (__bf16)(av[d] * inv);
}

// ---------------------------------------------------------------------------
// Block LayerNorm helper (blockDim.x == 256 == D)
// ---------------------------------------------------------------------------
__device__ __forceinline__ void ln_stats(float v, float* s1, float* s2,
                                         float& mean, float& inv) {
    const int t = threadIdx.x;
    s1[t] = v; s2[t] = v * v;
    __syncthreads();
    for (int w = 128; w > 0; w >>= 1) {
        if (t < w) { s1[t] += s1[t + w]; s2[t] += s2[t + w]; }
        __syncthreads();
    }
    mean = s1[0] * (1.f / 256.f);
    float var = s2[0] * (1.f / 256.f) - mean * mean;
    inv = rsqrtf(var + LNEPS);
    __syncthreads();
}

// scatter-by-perm residual + split LN1; writes f32 and bf16 copies
__global__ __launch_bounds__(256) void residual_ln1_kernel(
    const float* __restrict__ src_in, const float* __restrict__ src2f,
    const int* __restrict__ perm,
    const float* __restrict__ lg, const float* __restrict__ lb,
    const float* __restrict__ ig, const float* __restrict__ ib,
    float* __restrict__ x1f, __bf16* __restrict__ x1bf) {
    __shared__ float s1[256];
    __shared__ float s2[256];
    const int i = blockIdx.x, d = threadIdx.x;
    const int j = perm[i];
    const int r = (j / KSET) * KPAD + (j % KSET);
    float v = src_in[(size_t)i * DM + d] + src2f[(size_t)r * DM + d];
    float mean, inv;
    ln_stats(v, s1, s2, mean, inv);
    const float* g = (i < VNUM) ? lg : ig;
    const float* b = (i < VNUM) ? lb : ib;
    float o = (v - mean) * inv * g[d] + b[d];
    x1f[(size_t)i * DM + d] = o;
    x1bf[(size_t)i * DM + d] = (__bf16)o;
}

// split LN2, outer residual with layer input, final LN
__global__ __launch_bounds__(256) void ln2_final_kernel(
    const float* __restrict__ x2f, const float* __restrict__ src_in,
    const float* __restrict__ l2g, const float* __restrict__ l2b,
    const float* __restrict__ i2g, const float* __restrict__ i2b,
    const float* __restrict__ ng, const float* __restrict__ nb,
    float* __restrict__ out) {
    __shared__ float s1[256];
    __shared__ float s2[256];
    const int i = blockIdx.x, d = threadIdx.x;
    float v = x2f[(size_t)i * DM + d];
    float mean, inv;
    ln_stats(v, s1, s2, mean, inv);
    const float* g = (i < VNUM) ? l2g : i2g;
    const float* b = (i < VNUM) ? l2b : i2b;
    float t = (v - mean) * inv * g[d] + b[d];
    float u = t + src_in[(size_t)i * DM + d];
    ln_stats(u, s1, s2, mean, inv);
    out[(size_t)i * DM + d] = (u - mean) * inv * ng[d] + nb[d];
}

// ---------------------------------------------------------------------------
// Host side
// ---------------------------------------------------------------------------
static inline void launch_cvt(const float* in, __bf16* out, int n, hipStream_t st) {
    cvt_bf16_kernel<<<(n + 255) / 256, 256, 0, st>>>(in, out, n);
}

// mode: 0 = bf16 out; 1 = f32 out; 2 = relu+bf16 out; 3 = f32 out + addend
static inline void launch_gemm(int mode, const __bf16* A, const __bf16* W,
                               const float* bias, const float* addend, void* out,
                               int M, int Kd, int Nout, hipStream_t st) {
    const int Mtiles = M / 16;
    dim3 grid((Mtiles + 3) / 4, Nout / 64), block(128);
    switch (mode) {
        case 0: gemm_wmma_kernel<false, true,  false><<<grid, block, 0, st>>>(A, W, bias, nullptr, out, Mtiles, Kd, Nout); break;
        case 1: gemm_wmma_kernel<false, false, false><<<grid, block, 0, st>>>(A, W, bias, nullptr, out, Mtiles, Kd, Nout); break;
        case 2: gemm_wmma_kernel<true,  true,  false><<<grid, block, 0, st>>>(A, W, bias, nullptr, out, Mtiles, Kd, Nout); break;
        case 3: gemm_wmma_kernel<false, false, true ><<<grid, block, 0, st>>>(A, W, bias, addend, out, Mtiles, Kd, Nout); break;
    }
}

struct LayerWB {
    __bf16 *in_w, *out_w, *w1, *w2, *lw1, *lw2;
};

struct Scratch {
    int*    perm;
    __bf16 *qk, *feat, *qkp, *vp, *attn, *x1bf, *hbf;
    float  *src2f, *x1f, *x2f, *mid;
    LayerWB wb[2];
};

static void run_layer(const float* src_in, const float* pos, const int* inds,
                      const unsigned char* mask, const float* const* p,
                      const LayerWB& wb, const Scratch& S, float* out, hipStream_t st) {
    // p[]: 0 in_w, 1 in_b, 2 out_w, 3 out_b, 4 w1, 5 b1, 6 w2, 7 b2,
    //      8 lw1, 9 lb1, 10 lw2, 11 lb2, 12 n1g, 13 n1b, 14 n2g, 15 n2b,
    //      16 ln1g, 17 ln1b, 18 ln2g, 19 ln2b, 20 ng, 21 nb
    launch_cvt(p[0],  wb.in_w,  3 * DM * DM, st);
    launch_cvt(p[2],  wb.out_w, DM * DM, st);
    launch_cvt(p[4],  wb.w1,    DFFN * DM, st);
    launch_cvt(p[6],  wb.w2,    DM * DFFN, st);
    launch_cvt(p[8],  wb.lw1,   DFFN * DM, st);
    launch_cvt(p[10], wb.lw2,   DM * DFFN, st);

    perm_init_kernel<<<(NV + 255) / 256, 256, 0, st>>>(S.perm, NV);
    perm_min_kernel<<<(NSETS * KSET + 255) / 256, 256, 0, st>>>(inds, S.perm, NSETS * KSET);

    gather_kernel<<<MROWS, 256, 0, st>>>(src_in, pos, inds, S.qk, S.feat);

    // QK projection: Nout=512 (Wq|Wk rows of in_w), then V projection: Nout=256
    launch_gemm(0, S.qk,   wb.in_w,                 p[1],       nullptr, S.qkp, MROWS, DM, 512, st);
    launch_gemm(0, S.feat, wb.in_w + 2 * DM * DM,   p[1] + 512, nullptr, S.vp,  MROWS, DM, DM,  st);

    attention_kernel<<<dim3(NSETS, NH), KPAD, 0, st>>>(S.qkp, S.vp, mask, S.attn);

    // out projection -> f32 src2
    launch_gemm(1, S.attn, wb.out_w, p[3], nullptr, S.src2f, MROWS, DM, DM, st);

    residual_ln1_kernel<<<NV, 256, 0, st>>>(src_in, S.src2f, S.perm,
                                            p[16], p[17], p[12], p[13], S.x1f, S.x1bf);

    // FFN1 (relu), split lidar / image rows
    launch_gemm(2, S.x1bf,                       wb.lw1, p[9], nullptr, S.hbf,                        VNUM,       DM, DFFN, st);
    launch_gemm(2, S.x1bf + (size_t)VNUM * DM,   wb.w1,  p[5], nullptr, S.hbf + (size_t)VNUM * DFFN,  NV - VNUM,  DM, DFFN, st);

    // FFN2 + residual (addend = x1)
    launch_gemm(3, S.hbf,                        wb.lw2, p[11], S.x1f,                       S.x2f,                       VNUM,      DFFN, DM, st);
    launch_gemm(3, S.hbf + (size_t)VNUM * DFFN,  wb.w2,  p[7],  S.x1f + (size_t)VNUM * DM,   S.x2f + (size_t)VNUM * DM,   NV - VNUM, DFFN, DM, st);

    ln2_final_kernel<<<NV, 256, 0, st>>>(S.x2f, src_in,
                                         p[18], p[19], p[14], p[15], p[20], p[21], out);
}

extern "C" void kernel_launch(void* const* d_in, const int* in_sizes, int n_in,
                              void* d_out, int out_size, void* d_ws, size_t ws_size,
                              hipStream_t stream) {
    (void)in_sizes; (void)n_in; (void)out_size; (void)ws_size;

    // input order: src, pos0, pos1, inds0, inds1, mask0, mask1,
    //              params[0] (22 arrays), params[1] (22 arrays), voxel_num
    const float* src  = (const float*)d_in[0];
    const float* pos0 = (const float*)d_in[1];
    const float* pos1 = (const float*)d_in[2];
    const int*   inds0 = (const int*)d_in[3];
    const int*   inds1 = (const int*)d_in[4];
    const unsigned char* mask0 = (const unsigned char*)d_in[5];
    const unsigned char* mask1 = (const unsigned char*)d_in[6];
    const float* p0[22]; const float* p1[22];
    for (int t = 0; t < 22; ++t) p0[t] = (const float*)d_in[7 + t];
    for (int t = 0; t < 22; ++t) p1[t] = (const float*)d_in[29 + t];

    // workspace bump allocator
    char* base = (char*)d_ws;
    size_t off = 0;
    auto alloc = [&](size_t bytes) -> char* {
        char* ptr = base + off;
        off += (bytes + 255) & ~(size_t)255;
        return ptr;
    };

    Scratch S;
    S.perm  = (int*)   alloc((size_t)NV * 4);
    S.qk    = (__bf16*)alloc((size_t)MROWS * DM * 2);
    S.feat  = (__bf16*)alloc((size_t)MROWS * DM * 2);
    S.qkp   = (__bf16*)alloc((size_t)MROWS * 512 * 2);
    S.vp    = (__bf16*)alloc((size_t)MROWS * DM * 2);
    S.attn  = (__bf16*)alloc((size_t)MROWS * DM * 2);
    S.src2f = (float*) alloc((size_t)MROWS * DM * 4);
    S.x1f   = (float*) alloc((size_t)NV * DM * 4);
    S.x1bf  = (__bf16*)alloc((size_t)NV * DM * 2);
    S.hbf   = (__bf16*)alloc((size_t)NV * DFFN * 2);
    S.x2f   = (float*) alloc((size_t)NV * DM * 4);
    S.mid   = (float*) alloc((size_t)NV * DM * 4);
    for (int L = 0; L < 2; ++L) {
        S.wb[L].in_w  = (__bf16*)alloc((size_t)3 * DM * DM * 2);
        S.wb[L].out_w = (__bf16*)alloc((size_t)DM * DM * 2);
        S.wb[L].w1    = (__bf16*)alloc((size_t)DFFN * DM * 2);
        S.wb[L].w2    = (__bf16*)alloc((size_t)DM * DFFN * 2);
        S.wb[L].lw1   = (__bf16*)alloc((size_t)DFFN * DM * 2);
        S.wb[L].lw2   = (__bf16*)alloc((size_t)DM * DFFN * 2);
    }

    run_layer(src,   pos0, inds0, mask0, p0, S.wb[0], S, S.mid,         stream);
    run_layer(S.mid, pos1, inds1, mask1, p1, S.wb[1], S, (float*)d_out, stream);
}
